// MultiQueryAttention_35158602285389
// MI455X (gfx1250) — matrix operand order repaired and checked by reading, not compiled, not run on hardware
//
#include <hip/hip_runtime.h>
#include <hip/hip_bf16.h>

// ---------------------------------------------------------------------------
// Multi-Query Attention forward for MI455X (gfx1250), wave32 + WMMA bf16.
//   q = x Wq ; k = x Wk ; v = x Wv ; attn = softmax(mask(q k^T)/8) ; ctx=attn v
//   out = ctx Wo + b
// All GEMM work runs through v_wmma_f32_16x16x32_bf16 (fp32 accumulate).
// Attention K/V tiles staged to LDS via global_load_async_to_lds_b128
// (ASYNCcnt-tracked, s_wait_asynccnt fenced).
// ---------------------------------------------------------------------------

typedef __bf16 bf16_t;
typedef __attribute__((ext_vector_type(8)))  __bf16 v8bf;
typedef __attribute__((ext_vector_type(16))) __bf16 v16bf;
typedef __attribute__((ext_vector_type(8)))  float  v8f;

#define DEV __device__ __forceinline__

static constexpr int Bb  = 8;     // batch
static constexpr int S   = 1024;  // sequence
static constexpr int D   = 1024;  // model dim
static constexpr int H   = 16;    // heads
static constexpr int Dh  = 64;    // head dim

// ---- WMMA fragment helpers (layouts per CDNA5 ISA 7.12.2) -------

// A fragment 16x32 (MxK) from row-major source, base -> A[row0, k0].
// lane m (<16): K {0..7, 16..23}; lane m+16: K {8..15, 24..31}.
DEV v16bf load_frag_a(const bf16_t* base, int ld) {
  const int lane = threadIdx.x & 31;
  const int m = lane & 15, hi = lane >> 4;
  const bf16_t* p = base + (size_t)m * ld + hi * 8;
  union { v16bf v; v8bf h[2]; } u;
  u.h[0] = *(const v8bf*)p;
  u.h[1] = *(const v8bf*)(p + 16);
  return u.v;
}

// B fragment 32x16 (KxN) from pre-transposed Bt[N,K]; base -> Bt[col0, k0].
// lane n (<16): column n, K {0..7, 16..23}; lane n+16: column n, K {8..15, 24..31}.
DEV v16bf load_frag_b(const bf16_t* bt, int ld) {
  const int lane = threadIdx.x & 31;
  const int n = lane & 15, hi = lane >> 4;
  const bf16_t* p = bt + (size_t)n * ld + hi * 8;
  union { v16bf v; v8bf h[2]; } u;
  u.h[0] = *(const v8bf*)p;
  u.h[1] = *(const v8bf*)(p + 16);
  return u.v;
}

DEV v8f wmma_bf16(v16bf a, v16bf b, v8f c) {
  return __builtin_amdgcn_wmma_f32_16x16x32_bf16(
      /*neg_a=*/false, a, /*neg_b=*/false, b,
      /*c_mod=*/(short)0, c, /*reuse_a=*/false, /*reuse_b=*/false);
}

// 16-byte async copy global->LDS, one issue covers 32 lanes x 16B.
// ldsOff: per-lane LDS byte offset (added to the LDS base and the instruction offset).
DEV void async_copy_b128(unsigned int ldsOff, const void* gaddr) {
  asm volatile("global_load_async_to_lds_b128 %0, %1, off"
               :: "v"(ldsOff), "v"(gaddr) : "memory");
}
DEV void wait_asynccnt0() { asm volatile("s_wait_asynccnt 0x0" ::: "memory"); }
DEV void wait_dscnt0()    { asm volatile("s_wait_dscnt 0x0" ::: "memory"); }

// ---- stage 0: fp32 -> bf16 convert / transpose-convert ---------------------

__global__ void cvt_bf16_kernel(const float* __restrict__ in,
                                bf16_t* __restrict__ out, int n) {
  int i = blockIdx.x * blockDim.x + threadIdx.x;
  if (i < n) out[i] = (bf16_t)in[i];
}

// W[K,N] fp32 (row-major) -> Wt[N,K] bf16
__global__ void transpose_cvt_kernel(const float* __restrict__ W,
                                     bf16_t* __restrict__ Wt, int K, int N) {
  int i = blockIdx.x * blockDim.x + threadIdx.x;
  if (i >= K * N) return;
  int k = i / N, n = i % N;
  Wt[(size_t)n * K + k] = (bf16_t)W[i];
}

// ---- stage 1a: bf16 GEMM, 32x32 tile / wave (for narrow N = Dh) ------------
// vmode==1: scatter store for V as vT[b][d][s]  (row = b*S+s, col = d).

__global__ void gemm_bf16_kernel(const bf16_t* __restrict__ A,
                                 const bf16_t* __restrict__ Bt,
                                 bf16_t* __restrict__ C,
                                 int M, int N, int K, int vmode) {
  const int wave  = blockIdx.x * (blockDim.x >> 5) + (threadIdx.x >> 5);
  const int tilesN = N >> 5;
  const int tilesM = M >> 5;
  if (wave >= tilesM * tilesN) return;
  const int row0 = (wave / tilesN) * 32;
  const int col0 = (wave % tilesN) * 32;

  const bf16_t* a0p = A + (size_t)row0 * K;
  const bf16_t* a1p = A + (size_t)(row0 + 16) * K;
  const bf16_t* b0p = Bt + (size_t)col0 * K;
  const bf16_t* b1p = Bt + (size_t)(col0 + 16) * K;

  v8f acc00 = {}, acc01 = {}, acc10 = {}, acc11 = {};
  for (int k0 = 0; k0 < K; k0 += 32) {
    if (k0 + 32 < K) {
      __builtin_prefetch(a0p + k0 + 32, 0, 1);
      __builtin_prefetch(b0p + k0 + 32, 0, 1);
    }
    v16bf a0 = load_frag_a(a0p + k0, K);
    v16bf a1 = load_frag_a(a1p + k0, K);
    v16bf b0 = load_frag_b(b0p + k0, K);
    v16bf b1 = load_frag_b(b1p + k0, K);
    acc00 = wmma_bf16(a0, b0, acc00);
    acc01 = wmma_bf16(a0, b1, acc01);
    acc10 = wmma_bf16(a1, b0, acc10);
    acc11 = wmma_bf16(a1, b1, acc11);
  }

  const int lane = threadIdx.x & 31;
  const int nn = lane & 15, hi = lane >> 4;
#pragma unroll
  for (int r = 0; r < 8; ++r) {
#pragma unroll
    for (int t = 0; t < 4; ++t) {
      const v8f& acc = (t == 0) ? acc00 : (t == 1) ? acc01
                       : (t == 2) ? acc10 : acc11;
      int row = row0 + ((t >> 1) << 4) + r + hi * 8;
      int col = col0 + ((t & 1) << 4) + nn;
      size_t idx;
      if (vmode)   // vT[b][d][s]
        idx = ((size_t)(row >> 10) * Dh + col) * S + (row & (S - 1));
      else
        idx = (size_t)row * N + col;
      C[idx] = (bf16_t)acc[r];
    }
  }
}

// ---- stage 1b: bf16 GEMM, 64x64 tile / wave (16 WMMA per k-step) -----------
// OUTF32: fp32 store + bias (out-projection); else bf16 store.

template <bool OUTF32>
__global__ void gemm64_kernel(const bf16_t* __restrict__ A,
                              const bf16_t* __restrict__ Bt,
                              void* __restrict__ Cout,
                              const float* __restrict__ bias,
                              int M, int N, int K) {
  const int wave  = blockIdx.x * (blockDim.x >> 5) + (threadIdx.x >> 5);
  const int tilesN = N >> 6;
  const int tilesM = M >> 6;
  if (wave >= tilesM * tilesN) return;
  const int row0 = (wave / tilesN) * 64;
  const int col0 = (wave % tilesN) * 64;

  const bf16_t* ap[4];
  const bf16_t* bp[4];
#pragma unroll
  for (int i = 0; i < 4; ++i) {
    ap[i] = A + (size_t)(row0 + 16 * i) * K;
    bp[i] = Bt + (size_t)(col0 + 16 * i) * K;
  }

  const v8f vzero = {};
  v8f acc[16];
#pragma unroll
  for (int i = 0; i < 16; ++i) acc[i] = vzero;

  for (int k0 = 0; k0 < K; k0 += 32) {
    if (k0 + 32 < K) {
      __builtin_prefetch(ap[0] + k0 + 32, 0, 1);
      __builtin_prefetch(ap[2] + k0 + 32, 0, 1);
      __builtin_prefetch(bp[0] + k0 + 32, 0, 1);
      __builtin_prefetch(bp[2] + k0 + 32, 0, 1);
    }
    v16bf a[4], b[4];
#pragma unroll
    for (int i = 0; i < 4; ++i) a[i] = load_frag_a(ap[i] + k0, K);
#pragma unroll
    for (int i = 0; i < 4; ++i) b[i] = load_frag_b(bp[i] + k0, K);
#pragma unroll
    for (int i = 0; i < 4; ++i)
#pragma unroll
      for (int j = 0; j < 4; ++j)
        acc[i * 4 + j] = wmma_bf16(a[i], b[j], acc[i * 4 + j]);
  }

  const int lane = threadIdx.x & 31;
  const int nn = lane & 15, hi = lane >> 4;
#pragma unroll
  for (int i = 0; i < 4; ++i)
#pragma unroll
    for (int j = 0; j < 4; ++j)
#pragma unroll
      for (int r = 0; r < 8; ++r) {
        int row = row0 + 16 * i + r + hi * 8;
        int col = col0 + 16 * j + nn;
        if (OUTF32)
          ((float*)Cout)[(size_t)row * N + col] = acc[i * 4 + j][r] + bias[col];
        else
          ((bf16_t*)Cout)[(size_t)row * N + col] = (bf16_t)acc[i * 4 + j][r];
      }
}

// ---- stage 2: causal MQA attention (flash-style, one wave / 16 q rows) -----
// K/V tiles staged to LDS with async global->LDS copies (ASYNCcnt).

__global__ void __launch_bounds__(32)
mqa_attn_kernel(const bf16_t* __restrict__ qbf,   // [B*S, D] (head h at col h*64)
                const bf16_t* __restrict__ kbf,   // [B*S, Dh]
                const bf16_t* __restrict__ vT,    // [B][Dh][S]
                bf16_t* __restrict__ ctx) {       // [B*S, D]
  // Single shared block: P (16x32) | K-tile (32 keys x 64 d) | V-tile (64 d x 32 keys)
  __shared__ bf16_t smem[16 * 32 + 32 * 64 + 64 * 32];
  bf16_t* pLds = smem;                 //  512 elems
  bf16_t* kLds = smem + 16 * 32;       // 2048 elems (row = key, ld = 64)
  bf16_t* vLds = smem + 16 * 32 + 32 * 64; // 2048 elems (row = d, ld = 32)

  const int idx = blockIdx.x;
  const int qb = idx & 63;                        // q block (16 rows)
  const int h  = (idx >> 6) & 15;
  const int b  = idx >> 10;
  const int lane = threadIdx.x & 31;
  const int nn = lane & 15, hi = lane >> 4;

  // LDS byte offsets for async copies (flat LDS addr low 32 bits == LDS offset)
  const unsigned int kOff0 = (unsigned int)(uintptr_t)kLds + (unsigned int)lane * 16u;
  const unsigned int vOff0 = (unsigned int)(uintptr_t)vLds + (unsigned int)lane * 16u;
  const int dl = lane >> 2;            // d-row within a group of 8 (V copy)
  const int cb = (lane & 3) * 16;      // byte column within V row

  const bf16_t* qbase = qbf + ((size_t)(b * S + qb * 16)) * D + h * Dh;
  const v16bf qf0 = load_frag_a(qbase + 0,  D);   // d 0..31
  const v16bf qf1 = load_frag_a(qbase + 32, D);   // d 32..63

  v8f ctx0 = {}, ctx1 = {}, ctx2 = {}, ctx3 = {};
  float mrow[8], lrow[8];
#pragma unroll
  for (int r = 0; r < 8; ++r) { mrow[r] = -3.0e38f; lrow[r] = 0.0f; }

  const float scale = 0.125f;                     // 1/sqrt(64)
  const int nkb = ((qb * 16 + 15) >> 5) + 1;      // causal 32-key blocks

  for (int t = 0; t < nkb; ++t) {
    const int kk = t * 32;

    // ---- async stage K tile (4KB contiguous) and V tile (64 rows x 64B) ----
    wait_dscnt0();                     // drain prior LDS reads before overwrite
    const char* gk = (const char*)(kbf + ((size_t)(b * S + kk)) * Dh);
    const char* gv = (const char*)(vT + (size_t)b * Dh * S + kk);
#pragma unroll
    for (int i = 0; i < 8; ++i) {
      async_copy_b128(kOff0 + (unsigned)i * 512u, gk + i * 512 + lane * 16);
      async_copy_b128(vOff0 + (unsigned)i * 512u,
                      gv + (size_t)(i * 8 + dl) * (S * 2) + cb);
    }
    wait_asynccnt0();

    // ---- scores: A=q(16x64), B=k^T -> two 16x16 tiles, K split 32+32 -------
    v16bf kf00 = load_frag_b(kLds, Dh);                // keys kk..+15, d 0..31
    v16bf kf01 = load_frag_b(kLds + 32, Dh);           // keys kk..+15, d 32..63
    v16bf kf10 = load_frag_b(kLds + 16 * Dh, Dh);      // keys +16..+31
    v16bf kf11 = load_frag_b(kLds + 16 * Dh + 32, Dh);
    v8f s0 = {}, s1 = {};
    s0 = wmma_bf16(qf0, kf00, s0);
    s0 = wmma_bf16(qf1, kf01, s0);
    s1 = wmma_bf16(qf0, kf10, s1);
    s1 = wmma_bf16(qf1, kf11, s1);

    __syncthreads();   // protect pLds reuse across iterations (single wave)
#pragma unroll
    for (int r = 0; r < 8; ++r) {
      const int qrow = qb * 16 + r + hi * 8;
      const int kc0 = kk + nn, kc1 = kc0 + 16;
      float v0 = (kc0 <= qrow) ? s0[r] * scale : -3.0e38f;
      float v1 = (kc1 <= qrow) ? s1[r] * scale : -3.0e38f;
      float vm = fmaxf(v0, v1);
#pragma unroll
      for (int off = 1; off < 16; off <<= 1)
        vm = fmaxf(vm, __shfl_xor(vm, off, 16));
      const float mn = fmaxf(mrow[r], vm);
      const float corr = __expf(mrow[r] - mn);
      const float p0 = __expf(v0 - mn);
      const float p1 = __expf(v1 - mn);
      float ps = p0 + p1;
#pragma unroll
      for (int off = 1; off < 16; off <<= 1)
        ps += __shfl_xor(ps, off, 16);
      lrow[r] = lrow[r] * corr + ps;
      mrow[r] = mn;
      ctx0[r] *= corr; ctx1[r] *= corr; ctx2[r] *= corr; ctx3[r] *= corr;
      pLds[(r + hi * 8) * 32 + nn]      = (bf16_t)p0;
      pLds[(r + hi * 8) * 32 + 16 + nn] = (bf16_t)p1;
    }
    __syncthreads();   // P writes visible before A-layout reload

    const v16bf pf = load_frag_a(pLds, 32);        // P as 16x32 A operand
    v16bf vf0 = load_frag_b(vLds + 0  * 32, 32);   // d 0..15
    v16bf vf1 = load_frag_b(vLds + 16 * 32, 32);   // d 16..31
    v16bf vf2 = load_frag_b(vLds + 32 * 32, 32);   // d 32..47
    v16bf vf3 = load_frag_b(vLds + 48 * 32, 32);   // d 48..63
    ctx0 = wmma_bf16(pf, vf0, ctx0);
    ctx1 = wmma_bf16(pf, vf1, ctx1);
    ctx2 = wmma_bf16(pf, vf2, ctx2);
    ctx3 = wmma_bf16(pf, vf3, ctx3);
  }

  bf16_t* obase = ctx + ((size_t)(b * S + qb * 16)) * D + h * Dh;
#pragma unroll
  for (int r = 0; r < 8; ++r) {
    const float inv = 1.0f / lrow[r];
    const int row = r + hi * 8;
    obase[(size_t)row * D + 0  + nn] = (bf16_t)(ctx0[r] * inv);
    obase[(size_t)row * D + 16 + nn] = (bf16_t)(ctx1[r] * inv);
    obase[(size_t)row * D + 32 + nn] = (bf16_t)(ctx2[r] * inv);
    obase[(size_t)row * D + 48 + nn] = (bf16_t)(ctx3[r] * inv);
  }
}

// ---------------------------------------------------------------------------

extern "C" void kernel_launch(void* const* d_in, const int* in_sizes, int n_in,
                              void* d_out, int out_size, void* d_ws, size_t ws_size,
                              hipStream_t stream) {
  (void)in_sizes; (void)n_in; (void)out_size; (void)ws_size;
  const float* x  = (const float*)d_in[0];   // [8,1024,1024]
  const float* Wq = (const float*)d_in[1];   // [1024,1024]
  const float* Wk = (const float*)d_in[2];   // [1024,64]
  const float* Wv = (const float*)d_in[3];   // [1024,64]
  const float* Wo = (const float*)d_in[4];   // [1024,1024]
  const float* bo = (const float*)d_in[5];   // [1024]
  float* out = (float*)d_out;

  const int M = Bb * S;                      // 8192 rows
  bf16_t* ws = (bf16_t*)d_ws;
  size_t off = 0;
  bf16_t* xbf  = ws + off; off += (size_t)M * D;      // 8M
  bf16_t* WqT  = ws + off; off += (size_t)D * D;      // 1M
  bf16_t* WkT  = ws + off; off += (size_t)D * Dh;     // 64K
  bf16_t* WvT  = ws + off; off += (size_t)D * Dh;     // 64K
  bf16_t* WoT  = ws + off; off += (size_t)D * D;      // 1M
  bf16_t* qbf  = ws + off; off += (size_t)M * D;      // 8M
  bf16_t* kbf  = ws + off; off += (size_t)M * Dh;     // 512K
  bf16_t* vTb  = ws + off; off += (size_t)Bb * Dh * S;// 512K
  bf16_t* ctxb = ws + off; off += (size_t)M * D;      // 8M  (~55 MB bf16 total)

  // stage 0: conversions / weight transposes
  {
    int n = M * D;
    cvt_bf16_kernel<<<(n + 255) / 256, 256, 0, stream>>>(x, xbf, n);
    transpose_cvt_kernel<<<(D * D + 255) / 256, 256, 0, stream>>>(Wq, WqT, D, D);
    transpose_cvt_kernel<<<(D * Dh + 255) / 256, 256, 0, stream>>>(Wk, WkT, D, Dh);
    transpose_cvt_kernel<<<(D * Dh + 255) / 256, 256, 0, stream>>>(Wv, WvT, D, Dh);
    transpose_cvt_kernel<<<(D * D + 255) / 256, 256, 0, stream>>>(Wo, WoT, D, D);
  }

  // stage 1: projections (WMMA GEMM)
  {
    int waves_q = (M / 64) * (D / 64);       // 2048 waves, 64x64 tiles
    gemm64_kernel<false><<<(waves_q + 3) / 4, 128, 0, stream>>>(
        xbf, WqT, (void*)qbf, nullptr, M, D, D);
    int waves_kv = (M / 32) * (Dh / 32);     // 512 waves, 32x32 tiles
    gemm_bf16_kernel<<<(waves_kv + 3) / 4, 128, 0, stream>>>(
        xbf, WkT, kbf, M, Dh, D, 0);
    gemm_bf16_kernel<<<(waves_kv + 3) / 4, 128, 0, stream>>>(
        xbf, WvT, vTb, M, Dh, D, 1);         // transposed store -> vT[b][d][s]
  }

  // stage 2: attention — one wave per (b, h, q-tile of 16)
  mqa_attn_kernel<<<Bb * H * (S / 16), 32, 0, stream>>>(qbf, kbf, vTb, ctxb);

  // stage 3: output projection + bias (fp32 out, 64x64 tiles)
  {
    int waves_o = (M / 64) * (D / 64);
    gemm64_kernel<true><<<(waves_o + 3) / 4, 128, 0, stream>>>(
        ctxb, WoT, (void*)out, bo, M, D, D);
  }
}
